// ProteinGN_86981677678624
// MI455X (gfx1250) — matrix operand order repaired and checked by reading, not compiled, not run on hardware
//
#include <hip/hip_runtime.h>
#include <stdint.h>

// ---------------- problem constants (match reference) ----------------
#define NN 100000      // nodes
#define NE 1200000     // edges
#define NG 50          // graphs

typedef __attribute__((ext_vector_type(16))) __bf16 v16bf;
typedef __attribute__((ext_vector_type(8)))  float  v8f;
typedef __attribute__((ext_vector_type(4)))  int    vi4;

#if defined(__has_builtin)
# if __has_builtin(__builtin_amdgcn_global_load_async_to_lds_b128)
#  define HAVE_ASYNC 1
# endif
#endif
#ifndef HAVE_ASYNC
# define HAVE_ASYNC 0
#endif

__device__ __forceinline__ unsigned short f2bf(float f) {
  unsigned int u = __float_as_uint(f);
  unsigned int r = (u + 0x7FFFu + ((u >> 16) & 1u)) >> 16;   // RNE
  return (unsigned short)r;
}
__device__ __forceinline__ float bf2f(unsigned short h) {
  return __uint_as_float(((unsigned int)h) << 16);
}

union Frag16 { v16bf v; unsigned short u[16]; uint4 q[2]; };

// 16-byte global -> LDS copy; async (CDNA5 ASYNCcnt path) when available.
__device__ __forceinline__ void copy16_g2l(unsigned short* ldst, const unsigned short* gsrc) {
#if HAVE_ASYNC
  __builtin_amdgcn_global_load_async_to_lds_b128(
      (__attribute__((address_space(1))) vi4*)gsrc,
      (__attribute__((address_space(3))) vi4*)ldst, 0, 0);
#else
  *(uint4*)ldst = *(const uint4*)gsrc;
#endif
}

__device__ __forceinline__ void wait_async0() {
#if HAVE_ASYNC
# if __has_builtin(__builtin_amdgcn_s_wait_asynccnt)
  __builtin_amdgcn_s_wait_asynccnt(0);
# else
  asm volatile("s_wait_asynccnt 0x0" ::: "memory");
# endif
#endif
}

// ============================================================================
// One-time weight prep: fp32 [K,Nout] -> bf16 pre-swizzled B-fragment blob.
// Blob layout: [ct][kc][lane][16] with lane = (n&15) + 16*(khi>=16), elem=khi&15.
// ============================================================================
__global__ void prep_w_kernel(const float* __restrict__ src, unsigned short* __restrict__ dst,
                              int K, int Nout, int koff, int KC) {
  int i = blockIdx.x * blockDim.x + threadIdx.x;
  if (i >= K * Nout) return;
  const int k  = i / Nout;
  const int n  = i - k * Nout;
  const int kk = koff + k;
  const int kc = kk >> 5;
  const int kh = kk & 31;
  const int lane = (n & 15) + ((kh >> 4) << 4);
  const int elem = kh & 15;
  const int ct   = n >> 4;
  dst[(((size_t)(ct * KC + kc) * 32 + lane) << 4) + elem] = f2bf(src[i]);
}

// ============================================================================
// Fused linear: OUT[r,:] = act( sum_t A_t[idx_t(r),:] @ W_t + bias ), bf16 WMMA.
// A terms are bf16 (async-staged) unless flagged fp32 (float4 load + convert).
// Weights come pre-swizzled -> contiguous async copy to LDS.
// 256 threads = 8 waves; 128 rows/block; wave w owns rows w*16..+15.
// ============================================================================
__global__ __launch_bounds__(256) void wmma_linear_kernel(
    const void* __restrict__ A0, const int* __restrict__ I0, int K0,
    const void* __restrict__ A1, const int* __restrict__ I1, int K1,
    const void* __restrict__ A2, const int* __restrict__ I2, int K2,
    int fp32mask,
    const unsigned short* __restrict__ Wblob,
    const float* __restrict__ bias,
    unsigned short* __restrict__ OUT16, float* __restrict__ OUT32,
    int M, int Nout, int relu_flag)
{
  extern __shared__ unsigned short smem[];
  const int Ktot    = K0 + K1 + K2;
  const int KC      = Ktot >> 5;
  const int CT      = Nout >> 4;
  const int strideA = Ktot + 8;                     // mult of 8 -> 16B aligned rows
  unsigned short* sW = smem;                        // [CT][KC][32][16]
  unsigned short* sA = smem + (size_t)Ktot * Nout;  // [128][strideA]

  const int tid      = threadIdx.x;
  const int blockRow = blockIdx.x * 128;

  // ---- weights: contiguous copy of pre-swizzled blob ----
  {
    const int total8 = (Ktot * Nout) >> 3;
    for (int i = tid; i < total8; i += 256)
      copy16_g2l(sW + ((size_t)i << 3), Wblob + ((size_t)i << 3));
  }

  // ---- activations (optional gather) ----
  {
    const void* As[3] = {A0, A1, A2};
    const int*  Is[3] = {I0, I1, I2};
    const int   Ks[3] = {K0, K1, K2};
    int koff = 0;
    for (int t = 0; t < 3; ++t) {
      const int K = Ks[t];
      if (As[t] && K > 0) {
        if (fp32mask & (1 << t)) {
          const float* A  = (const float*)As[t];
          const int nch   = K >> 2;                 // 4 floats per chunk
          const int total = 128 * nch;
          for (int i = tid; i < total; i += 256) {
            const int r  = i / nch;
            const int c  = i - r * nch;
            const int gr = blockRow + r;
            unsigned short* dst = sA + (size_t)r * strideA + koff + (c << 2);
            if (gr < M) {
              const int row  = Is[t] ? Is[t][gr] : gr;
              const float4 v = *(const float4*)(A + (size_t)row * K + (c << 2));
              ushort4 h = make_ushort4(f2bf(v.x), f2bf(v.y), f2bf(v.z), f2bf(v.w));
              *(ushort4*)dst = h;
            } else {
              *(ushort4*)dst = make_ushort4(0, 0, 0, 0);
            }
          }
        } else {
          const unsigned short* A = (const unsigned short*)As[t];
          const int nch   = K >> 3;                 // 8 bf16 per chunk (16B)
          const int total = 128 * nch;
          for (int i = tid; i < total; i += 256) {
            const int r  = i / nch;
            const int c  = i - r * nch;
            const int gr = blockRow + r;
            unsigned short* dst = sA + (size_t)r * strideA + koff + (c << 3);
            if (gr < M) {
              const int row = Is[t] ? Is[t][gr] : gr;
              copy16_g2l(dst, A + (size_t)row * K + (c << 3));
            } else {
              *(uint4*)dst = make_uint4(0, 0, 0, 0);
            }
          }
        }
      }
      koff += K;
    }
  }
  wait_async0();
  __syncthreads();

  // ---- WMMA compute ----
  const int wave    = tid >> 5;
  const int lane    = tid & 31;
  const int half    = lane >> 4;
  const int lm      = lane & 15;
  const int rowBase = wave * 16;

  for (int ct = 0; ct < CT; ++ct) {
    v8f acc = {};
    for (int kc = 0; kc < KC; ++kc) {
      Frag16 a, b;
      const unsigned short* ap =
          sA + (size_t)(rowBase + lm) * strideA + (kc << 5) + (half << 4);
      a.q[0] = *(const uint4*)(ap);
      a.q[1] = *(const uint4*)(ap + 8);
      const unsigned short* bp =
          sW + (((size_t)(ct * KC + kc) * 32 + lane) << 4);
      b.q[0] = *(const uint4*)(bp);
      b.q[1] = *(const uint4*)(bp + 8);
      acc = __builtin_amdgcn_wmma_f32_16x16x32_bf16(
          false, a.v, false, b.v, (short)0, acc, false, false);
    }
    const int   n  = (ct << 4) + lm;
    const float bv = bias ? bias[n] : 0.0f;
#pragma unroll
    for (int i = 0; i < 8; ++i) {
      const int m  = i + half * 8;
      const int gr = blockRow + rowBase + m;
      if (gr < M) {
        float v = acc[i] + bv;
        if (relu_flag) v = fmaxf(v, 0.0f);
        const size_t idx = (size_t)gr * Nout + n;
        if (OUT16) OUT16[idx] = f2bf(v);
        else       OUT32[idx] = v;
      }
    }
  }
}

// ---------------------------- helper kernels --------------------------------
__global__ void fill_kernel(float* p, size_t n, float v) {
  size_t i = (size_t)blockIdx.x * blockDim.x + threadIdx.x;
  if (i < n) p[i] = v;
}

__global__ void seg_scatter_bf16_kernel(const unsigned short* __restrict__ X,
                                        const int* __restrict__ seg,
                                        float* __restrict__ sum, float* __restrict__ cnt,
                                        int rows, int C) {
  size_t i = (size_t)blockIdx.x * blockDim.x + threadIdx.x;
  size_t total = (size_t)rows * (size_t)C;
  if (i >= total) return;
  const int r = (int)(i / C);
  const int c = (int)(i - (size_t)r * C);
  const int s = seg[r];
  atomicAdd(&sum[(size_t)s * C + c], bf2f(X[i]));
  if (c == 0) atomicAdd(&cnt[s], 1.0f);
}

__global__ void seg_div_kernel(float* sum, const float* __restrict__ cnt, int segs, int C) {
  int i = blockIdx.x * blockDim.x + threadIdx.x;
  if (i >= segs * C) return;
  const float c = cnt[i / C];
  sum[i] = (c > 0.0f) ? sum[i] / c : 0.0f;
}

__global__ void seg_div_bf16_kernel(const float* __restrict__ sum, const float* __restrict__ cnt,
                                    unsigned short* __restrict__ dst, int segs, int C) {
  size_t i = (size_t)blockIdx.x * blockDim.x + threadIdx.x;
  if (i >= (size_t)segs * C) return;
  const float c = cnt[i / C];
  dst[i] = f2bf((c > 0.0f) ? sum[i] / c : 0.0f);
}

__global__ void add_bf16_kernel(unsigned short* x, const unsigned short* __restrict__ y, size_t n) {
  size_t i = (size_t)blockIdx.x * blockDim.x + threadIdx.x;
  if (i < n) x[i] = f2bf(bf2f(x[i]) + bf2f(y[i]));
}

__global__ void f2bf_kernel(const float* __restrict__ src, unsigned short* __restrict__ dst, int n) {
  int i = blockIdx.x * blockDim.x + threadIdx.x;
  if (i < n) dst[i] = f2bf(src[i]);
}

// tiny [G, Nout] linear (G=50): plain VALU, not worth WMMA
__global__ void g_linear3_kernel(const float* A0, int K0, const float* W0,
                                 const float* A1, int K1, const float* W1,
                                 const float* A2, int K2, const float* W2,
                                 const float* bias, const float* resid,
                                 float* OUT, int segs, int Nout, int relu_flag) {
  int i = blockIdx.x * blockDim.x + threadIdx.x;
  if (i >= segs * Nout) return;
  const int g = i / Nout, n = i - g * Nout;
  float acc = bias ? bias[n] : 0.0f;
  if (A0) for (int k = 0; k < K0; ++k) acc += A0[g * K0 + k] * W0[k * Nout + n];
  if (A1) for (int k = 0; k < K1; ++k) acc += A1[g * K1 + k] * W1[k * Nout + n];
  if (A2) for (int k = 0; k < K2; ++k) acc += A2[g * K2 + k] * W2[k * Nout + n];
  if (relu_flag) acc = fmaxf(acc, 0.0f);
  if (resid) acc += resid[i];
  OUT[i] = acc;
}

__global__ void readout_bf16_kernel(const unsigned short* __restrict__ X,
                                    const float* __restrict__ W, const float* __restrict__ b,
                                    float* __restrict__ out, int M, int K) {
  int i = blockIdx.x * blockDim.x + threadIdx.x;
  if (i >= M) return;
  float acc = b[0];
  const unsigned short* x = X + (size_t)i * K;
  for (int k = 0; k < K; ++k) acc += bf2f(x[k]) * W[k];
  out[i] = 1.0f / (1.0f + __expf(-acc));
}

__global__ void readout_f32_kernel(const float* __restrict__ X,
                                   const float* __restrict__ W, const float* __restrict__ b,
                                   float* __restrict__ out, int M, int K) {
  int i = blockIdx.x * blockDim.x + threadIdx.x;
  if (i >= M) return;
  float acc = b[0];
  const float* x = X + (size_t)i * K;
  for (int k = 0; k < K; ++k) acc += x[k] * W[k];
  out[i] = 1.0f / (1.0f + __expf(-acc));
}

// ---------------------------------------------------------------------------
extern "C" void kernel_launch(void* const* d_in, const int* in_sizes, int n_in,
                              void* d_out, int out_size, void* d_ws, size_t ws_size,
                              hipStream_t stream) {
  (void)in_sizes; (void)n_in; (void)out_size; (void)ws_size;

  const float* nodes      = (const float*)d_in[0];
  const float* edges      = (const float*)d_in[1];
  const int*   senders    = (const int*)d_in[2];
  const int*   receivers  = (const int*)d_in[3];
  const int*   node_graph = (const int*)d_in[4];
  const int*   edge_graph = (const int*)d_in[5];
  int pi = 6;
  const float* e1_W = (const float*)d_in[pi++]; const float* e1_b = (const float*)d_in[pi++];
  const float* n1_W = (const float*)d_in[pi++]; const float* n1_b = (const float*)d_in[pi++];
  const float* e2_W = (const float*)d_in[pi++]; const float* e2_b = (const float*)d_in[pi++];
  const float* n2_W = (const float*)d_in[pi++]; const float* n2_b = (const float*)d_in[pi++];
  const float* g2_Wn = (const float*)d_in[pi++]; const float* g2_b = (const float*)d_in[pi++];
  const float* he_We = (const float*)d_in[pi++]; const float* he_Ws = (const float*)d_in[pi++];
  const float* he_Wg = (const float*)d_in[pi++]; const float* he_b  = (const float*)d_in[pi++];
  const float* hn_Wn = (const float*)d_in[pi++]; const float* hn_Wi = (const float*)d_in[pi++];
  const float* hn_Wg = (const float*)d_in[pi++]; const float* hn_b  = (const float*)d_in[pi++];
  const float* hg_We = (const float*)d_in[pi++]; const float* hg_Wn = (const float*)d_in[pi++];
  const float* hg_Wg = (const float*)d_in[pi++]; const float* hg_b  = (const float*)d_in[pi++];
  const float* ro_n_W = (const float*)d_in[pi++]; const float* ro_n_b = (const float*)d_in[pi++];
  const float* ro_g_W = (const float*)d_in[pi++]; const float* ro_g_b = (const float*)d_in[pi++];

  float* out = (float*)d_out;

  // -------- workspace layout --------
  unsigned short* wsu = (unsigned short*)d_ws;
  // pre-swizzled weight blobs (bf16)
  unsigned short* wb_e1 = wsu + 0;        // 64*32   = 2048
  unsigned short* wb_n1 = wsu + 2048;     // 128*256 = 32768
  unsigned short* wb_e2 = wsu + 34816;    // 32*64   = 2048
  unsigned short* wb_n2 = wsu + 36864;    // 256*128 = 32768
  unsigned short* wb_he = wsu + 69632;    // 224*64  = 14336
  unsigned short* wb_hn = wsu + 83968;    // 224*128 = 28672 -> end 112640
  // bf16 activations
  unsigned short* e_buf16 = wsu + 112640;                      // E*64
  unsigned short* n_buf16 = e_buf16 + (size_t)NE * 64;         // N*128
  unsigned short* e_tmp16 = n_buf16 + (size_t)NN * 128;        // E*64 (also e1/n1 staging)
  unsigned short* n_tmp16 = e_tmp16 + (size_t)NE * 64;         // N*128
  unsigned short* inc16   = n_tmp16 + (size_t)NN * 128;        // N*64
  unsigned short* g16A    = inc16 + (size_t)NN * 64;           // G*32
  unsigned short* g16B    = g16A + NG * 32;                    // G*32
  unsigned short* e1_16   = e_tmp16;                           // E*32 (pre-hop)
  unsigned short* n1_16   = e_tmp16 + (size_t)NE * 32;         // N*256 (pre-hop)
  // fp32 region
  size_t ucount = (size_t)((g16B + NG * 32) - wsu);
  if (ucount & 1) ++ucount;
  float* inc   = (float*)(wsu + ucount);                       // N*64
  float* icnt  = inc + (size_t)NN * 64;                        // N
  float* gA    = icnt + NN;                                    // G*32
  float* gB    = gA + NG * 32;                                 // G*32
  float* eg    = gB + NG * 32;                                 // G*64
  float* ng    = eg + NG * 64;                                 // G*128
  float* cntA  = ng + NG * 128;                                // G
  float* cntB  = cntA + NG;                                    // G

  auto prepw = [&](const float* src, unsigned short* dst, int K, int Nout, int koff, int KC) {
    prep_w_kernel<<<(K * Nout + 255) / 256, 256, 0, stream>>>(src, dst, K, Nout, koff, KC);
  };
  auto gemm = [&](const void* A0, const int* I0, int K0,
                  const void* A1, const int* I1, int K1,
                  const void* A2, const int* I2, int K2,
                  int fp32mask, const unsigned short* blob, const float* bias,
                  unsigned short* O16, int M, int Nout, int relu) {
    const int Ktot = K0 + K1 + K2;
    const size_t lds = ((size_t)Ktot * Nout + 128 * (size_t)(Ktot + 8)) * sizeof(unsigned short);
    wmma_linear_kernel<<<dim3((M + 127) / 128), 256, lds, stream>>>(
        A0, I0, K0, A1, I1, K1, A2, I2, K2, fp32mask, blob, bias, O16, nullptr, M, Nout, relu);
  };
  auto fill = [&](float* p, size_t n, float v) {
    fill_kernel<<<(unsigned)((n + 255) / 256), 256, 0, stream>>>(p, n, v);
  };
  auto scat = [&](const unsigned short* X, const int* seg, float* sum, float* cnt, int rows, int C) {
    size_t total = (size_t)rows * C;
    seg_scatter_bf16_kernel<<<(unsigned)((total + 255) / 256), 256, 0, stream>>>(X, seg, sum, cnt, rows, C);
  };
  auto segdiv = [&](float* sum, const float* cnt, int segs, int C) {
    seg_div_kernel<<<(unsigned)(((size_t)segs * C + 255) / 256), 256, 0, stream>>>(sum, cnt, segs, C);
  };

  const void* NZ = nullptr; const int* NI = nullptr; const float* NF32 = nullptr;

  // ---- one-time weight prep (pre-swizzled bf16 blobs) ----
  prepw(e1_W, wb_e1, 64, 32, 0, 2);
  prepw(n1_W, wb_n1, 128, 256, 0, 4);
  prepw(e2_W, wb_e2, 32, 64, 0, 1);
  prepw(n2_W, wb_n2, 256, 128, 0, 8);
  prepw(he_We, wb_he, 64, 64, 0, 7);
  prepw(he_Ws, wb_he, 128, 64, 64, 7);
  prepw(he_Wg, wb_he, 32, 64, 192, 7);
  prepw(hn_Wn, wb_hn, 128, 128, 0, 7);
  prepw(hn_Wi, wb_hn, 64, 128, 128, 7);
  prepw(hn_Wg, wb_hn, 32, 128, 192, 7);

  // ---- encoder ----
  gemm(edges, NI, 64,  NZ, NI, 0, NZ, NI, 0, 1, wb_e1, e1_b, e1_16, NE, 32, 1);
  gemm(nodes, NI, 128, NZ, NI, 0, NZ, NI, 0, 1, wb_n1, n1_b, n1_16, NN, 256, 1);
  gemm(e1_16, NI, 32,  NZ, NI, 0, NZ, NI, 0, 0, wb_e2, e2_b, e_buf16, NE, 64, 1);
  gemm(n1_16, NI, 256, NZ, NI, 0, NZ, NI, 0, 0, wb_n2, n2_b, n_buf16, NN, 128, 1);
  // g = seg_mean(n) @ g2_Wn + b
  fill(ng, (size_t)NG * 128, 0.0f); fill(cntA, NG, 0.0f);
  scat(n_buf16, node_graph, ng, cntA, NN, 128);
  segdiv(ng, cntA, NG, 128);
  g_linear3_kernel<<<(NG * 32 + 255) / 256, 256, 0, stream>>>(
      ng, 128, g2_Wn, NF32, 0, NF32, NF32, 0, NF32, g2_b, nullptr, gA, NG, 32, 0);
  f2bf_kernel<<<(NG * 32 + 255) / 256, 256, 0, stream>>>(gA, g16A, NG * 32);

  float* g_cur = gA; float* g_nxt = gB;
  unsigned short* g16cur = g16A; unsigned short* g16nxt = g16B;

  // ---- residual message-passing hops (shared weights) ----
  for (int hop = 0; hop < 2; ++hop) {
    // e2 = relu(e@We + n[senders]@Ws + g[edge_graph]@Wg + b)
    gemm(e_buf16, NI, 64, n_buf16, senders, 128, g16cur, edge_graph, 32,
         0, wb_he, he_b, e_tmp16, NE, 64, 1);
    // inc = seg_mean(e2, receivers, N)
    fill(inc, (size_t)NN * 64, 0.0f); fill(icnt, NN, 0.0f);
    scat(e_tmp16, receivers, inc, icnt, NE, 64);
    seg_div_bf16_kernel<<<(unsigned)(((size_t)NN * 64 + 255) / 256), 256, 0, stream>>>(
        inc, icnt, inc16, NN, 64);
    // n2 = relu(n@Wn + inc@Wi + g[node_graph]@Wg + b)
    gemm(n_buf16, NI, 128, inc16, NI, 64, g16cur, node_graph, 32,
         0, wb_hn, hn_b, n_tmp16, NN, 128, 1);
    // g2 = relu(mean_e(e2)@We + mean_n(n2)@Wn + g@Wg + b); g_next = g + g2
    fill(eg, (size_t)NG * 64, 0.0f);  fill(cntA, NG, 0.0f);
    fill(ng, (size_t)NG * 128, 0.0f); fill(cntB, NG, 0.0f);
    scat(e_tmp16, edge_graph, eg, cntA, NE, 64);
    scat(n_tmp16, node_graph, ng, cntB, NN, 128);
    segdiv(eg, cntA, NG, 64);
    segdiv(ng, cntB, NG, 128);
    g_linear3_kernel<<<(NG * 32 + 255) / 256, 256, 0, stream>>>(
        eg, 64, hg_We, ng, 128, hg_Wn, g_cur, 32, hg_Wg, hg_b, g_cur, g_nxt, NG, 32, 1);
    f2bf_kernel<<<(NG * 32 + 255) / 256, 256, 0, stream>>>(g_nxt, g16nxt, NG * 32);
    // residuals
    add_bf16_kernel<<<(unsigned)(((size_t)NE * 64 + 255) / 256), 256, 0, stream>>>(
        e_buf16, e_tmp16, (size_t)NE * 64);
    add_bf16_kernel<<<(unsigned)(((size_t)NN * 128 + 255) / 256), 256, 0, stream>>>(
        n_buf16, n_tmp16, (size_t)NN * 128);
    { float* t = g_cur; g_cur = g_nxt; g_nxt = t; }
    { unsigned short* t = g16cur; g16cur = g16nxt; g16nxt = t; }
  }

  // ---- readouts ----
  readout_bf16_kernel<<<(NN + 255) / 256, 256, 0, stream>>>(n_buf16, ro_n_W, ro_n_b, out, NN, 128);
  readout_f32_kernel<<<1, 64, 0, stream>>>(g_cur, ro_g_W, ro_g_b, out + NN, NG, 32);
}
